// TDID_30889404793293
// MI455X (gfx1250) — matrix operand order repaired and checked by reading, not compiled, not run on hardware
//
#include <hip/hip_runtime.h>
#include <hip/hip_bf16.h>

// ---------------------------------------------------------------------------
// TDID head network on MI455X (gfx1250).
// Compute-bound (~850 GFLOP vs ~1GB HBM traffic): 3x3 convs run as implicit
// GEMM on v_wmma_f32_16x16x32_bf16. LDS-staged block GEMM: 128x128 tile per
// 8-wave workgroup, double-buffered; B staged with CDNA5 async-to-LDS ops
// (ASYNCcnt), A staged global->VGPR->ds_store with 3x3 halo zero-fill.
// ---------------------------------------------------------------------------

typedef __attribute__((ext_vector_type(16))) __bf16 v16bf;
typedef __attribute__((ext_vector_type(8)))  float  v8f;

#define Hh 48
#define Ww 64
#define PIX 3072              // 48*64
#define BATCH 8
#define MTOT (BATCH * PIX)    // 24576 pixels (GEMM M)
#define A_STRIDE 40           // padded LDS row stride (elems) for 32-K A rows

__device__ __forceinline__ __bf16 f2b(float x) { return (__bf16)x; }

// ---------------- kernels ----------------

// 6x6 spatial max-pool of target features -> t1p/t2p (8,512)
__global__ void pool_kernel(const float* __restrict__ tf,
                            float* __restrict__ t1p, float* __restrict__ t2p) {
    int idx = blockIdx.x * blockDim.x + threadIdx.x;
    if (idx >= BATCH * 512) return;
    int n = idx / 512, c = idx % 512;
    const float* p1 = tf + ((size_t)(2 * n) * 512 + c) * 36;
    const float* p2 = tf + ((size_t)(2 * n + 1) * 512 + c) * 36;
    float m1 = -3.4e38f, m2 = -3.4e38f;
#pragma unroll
    for (int i = 0; i < 36; ++i) {
        m1 = fmaxf(m1, p1[i]);
        m2 = fmaxf(m2, p2[i]);
    }
    t1p[idx] = m1;
    t2p[idx] = m2;
}

// Weight transform: (OC, IC, taps) f32 -> WMMA fragment-major bf16.
// K ordering: k = tap*IC + ic. Fragment = 32K x 16N tile, 512 elems, indexed
// [lane*16 + j] with k = kt*32 + (lane>>4)*16 + j, n = nt*16 + (lane&15).
__global__ void wtrans_kernel(const float* __restrict__ w, __bf16* __restrict__ wt,
                              int OC, int IC, int taps) {
    size_t i = (size_t)blockIdx.x * blockDim.x + threadIdx.x;
    size_t total = (size_t)OC * IC * taps;
    if (i >= total) return;
    const int j    = (int)(i & 15);
    const int lane = (int)((i >> 4) & 31);
    const size_t f = i >> 9;
    const int NT = OC >> 4;
    const int nt = (int)(f % NT);
    const size_t kt = f / NT;
    const int k  = (int)(kt * 32) + ((lane >> 4) << 4) + j;
    const int nn = (nt << 4) + (lane & 15);
    const int tap = k / IC;
    const int ic  = k % IC;
    wt[i] = f2b(w[((size_t)nn * IC + ic) * taps + tap]);
}

// Fused: img NCHW f32 -> cat[0:512] NHWC bf16; diffs (img - pooled targets)
// -> dif NHWC bf16 (1024 ch); depthwise 6x6 cross-correlation -> ccb NHWC bf16.
__global__ void corr_diff_kernel(const float* __restrict__ img, const float* __restrict__ tf,
                                 const float* __restrict__ t1p, const float* __restrict__ t2p,
                                 __bf16* __restrict__ cat, __bf16* __restrict__ dif,
                                 __bf16* __restrict__ ccb) {
    size_t i = (size_t)blockIdx.x * blockDim.x + threadIdx.x;
    const size_t total = (size_t)BATCH * 512 * PIX;
    if (i >= total) return;
    const int x = (int)(i & 63);
    const int y = (int)((i >> 6) % Hh);
    const int c = (int)((i / PIX) & 511);
    const int n = (int)(i / ((size_t)PIX * 512));
    const float* ip = img + ((size_t)(n * 512 + c)) * PIX;
    const float v = ip[y * Ww + x];
    const size_t pix = (size_t)(n * Hh + y) * Ww + x;
    cat[pix * 1024 + c]       = f2b(v);
    dif[pix * 1024 + c]       = f2b(v - t1p[n * 512 + c]);
    dif[pix * 1024 + 512 + c] = f2b(v - t2p[n * 512 + c]);
    const float* k1 = tf + ((size_t)(2 * n) * 512 + c) * 36;
    const float* k2 = tf + ((size_t)(2 * n + 1) * 512 + c) * 36;
    float s1 = 0.f, s2 = 0.f;
#pragma unroll
    for (int dy = 0; dy < 6; ++dy) {
        const int yy = y - 3 + dy;
        if (yy < 0 || yy >= Hh) continue;
#pragma unroll
        for (int dx = 0; dx < 6; ++dx) {
            const int xx = x - 3 + dx;
            if (xx < 0 || xx >= Ww) continue;
            const float iv = ip[yy * Ww + xx];
            s1 += iv * k1[dy * 6 + dx];
            s2 += iv * k2[dy * 6 + dx];
        }
    }
    ccb[pix * 1024 + c]       = f2b(s1);
    ccb[pix * 1024 + 512 + c] = f2b(s2);
}

// LDS-staged implicit-GEMM conv (1x1 when is3==0, 3x3 pad-1 when is3==1).
// Block: 256 thr = 8 waves, tile 128(M pixels) x 128(OC). Wave (mw 0..3,
// nw 0..1) computes 32x64 -> 8 f32 accumulators. Double-buffered LDS.
__global__ void __launch_bounds__(256)
conv_gemm_kernel(const __bf16* __restrict__ act, const __bf16* __restrict__ wf,
                 const float* __restrict__ bias, __bf16* __restrict__ out,
                 int IC, int OC, int is3, int outLD, int outOff, int relu) {
    __shared__ __align__(16) __bf16 As[2][128 * A_STRIDE]; // 128 rows x 32K (+pad)
    __shared__ __align__(16) __bf16 Bs[2][4096];           // 8 fragments x 512

    const int tid  = threadIdx.x;
    const int lane = tid & 31;
    const int wid  = tid >> 5;
    const int mw   = wid >> 1;   // 0..3 : 32-pixel M strip
    const int nw   = wid & 1;    // 0..1 : 64-channel N strip

    const int NB = OC >> 7;                 // 128-wide N blocks
    const int bn = blockIdx.x % NB;
    const int bm = blockIdx.x / NB;
    const int m0 = bm << 7;                 // 128 pixels = 2 full image rows
    const int n0 = bn << 7;
    const int NT  = OC >> 4;
    const int ntb = n0 >> 4;

    const int nimg = m0 / PIX;
    const int y0   = (m0 % PIX) >> 6;       // block covers rows y0, y0+1

    const int ICsteps = IC >> 5;
    const int KSTEPS  = (is3 ? 9 : 1) * ICsteps;

    // staging thread mapping: row smi (0..127), K half sh (0..1)
    const int smi = tid >> 1;
    const int sh  = tid & 1;

    v8f acc[2][4];
#pragma unroll
    for (int mi = 0; mi < 2; ++mi)
#pragma unroll
        for (int ni = 0; ni < 4; ++ni) acc[mi][ni] = {};

    // --- staging helpers ---
    // B: one contiguous 8KB block per 32-K step -> async copy to LDS (ASYNCcnt)
    auto stageB = [&](int ks, int buf) {
        const __bf16* src = wf + (((size_t)ks * NT + ntb) << 9) + ((size_t)tid << 4);
        unsigned long long g0 = (unsigned long long)(uintptr_t)src;
        unsigned int l0 = (unsigned int)(uintptr_t)(&Bs[buf][tid << 4]);
        asm volatile("global_load_async_to_lds_b128 %0, %1, off"
                     :: "v"(l0), "v"(g0) : "memory");
        asm volatile("global_load_async_to_lds_b128 %0, %1, off offset:16"
                     :: "v"(l0), "v"(g0) : "memory");
    };
    // A: global -> VGPR (halo zero-fill), later ds_store into LDS
    auto loadA = [&](int ks, uint4& r0, uint4& r1) {
        const int tap = ks / ICsteps;
        const int kc  = (ks % ICsteps) << 5;
        const int dy  = is3 ? (tap / 3 - 1) : 0;
        const int dx  = is3 ? (tap % 3 - 1) : 0;
        const int yy  = y0 + (smi >> 6) + dy;
        const int xx  = (smi & 63) + dx;
        r0 = make_uint4(0u, 0u, 0u, 0u);
        r1 = make_uint4(0u, 0u, 0u, 0u);
        if (yy >= 0 && yy < Hh && xx >= 0 && xx < Ww) {
            const uint4* gp = (const uint4*)(act +
                (((size_t)nimg * Hh + yy) * Ww + xx) * IC + kc + (sh << 4));
            r0 = gp[0];
            r1 = gp[1];
        }
    };
    auto storeA = [&](int buf, uint4 r0, uint4 r1) {
        uint4* dp = (uint4*)&As[buf][smi * A_STRIDE + (sh << 4)];
        dp[0] = r0;
        dp[1] = r1;
    };

    // --- prologue: stage step 0 into buffer 0 ---
    uint4 ar0, ar1;
    stageB(0, 0);
    loadA(0, ar0, ar1);
    storeA(0, ar0, ar1);
    asm volatile("s_wait_asynccnt 0x0" ::: "memory");
    __syncthreads();

    // --- main pipeline ---
    for (int ks = 0; ks < KSTEPS; ++ks) {
        const int buf  = ks & 1;
        const int nbuf = buf ^ 1;
        const bool more = (ks + 1) < KSTEPS;
        if (more) {
            stageB(ks + 1, nbuf);       // async, overlaps with WMMA below
            loadA(ks + 1, ar0, ar1);    // global loads in flight during WMMA
        }

        // fragments from LDS
        v16bf a[2], b[4];
#pragma unroll
        for (int mi = 0; mi < 2; ++mi) {
            const __bf16* ap = &As[buf][(mw * 32 + mi * 16 + (lane & 15)) * A_STRIDE
                                        + ((lane >> 4) * 16)];
#pragma unroll
            for (int j = 0; j < 8; ++j) { a[mi][j] = ap[j]; a[mi][j + 8] = ap[8 + j]; }
        }
#pragma unroll
        for (int ni = 0; ni < 4; ++ni) {
            const __bf16* bp = &Bs[buf][(nw * 4 + ni) * 512 + lane * 16];
#pragma unroll
            for (int j = 0; j < 16; ++j) b[ni][j] = bp[j];
        }
#pragma unroll
        for (int mi = 0; mi < 2; ++mi)
#pragma unroll
            for (int ni = 0; ni < 4; ++ni)
                acc[mi][ni] = __builtin_amdgcn_wmma_f32_16x16x32_bf16(
                    false, a[mi], false, b[ni], (short)0, acc[mi][ni], false, false);

        if (more) {
            storeA(nbuf, ar0, ar1);
            asm volatile("s_wait_asynccnt 0x0" ::: "memory");
        }
        __syncthreads();
    }

    // --- epilogue: C/D layout row m = (lane>>4)*8 + j, col n = lane&15 ---
    const int lr = (lane >> 4) * 8;
    const int lc = lane & 15;
#pragma unroll
    for (int mi = 0; mi < 2; ++mi) {
#pragma unroll
        for (int ni = 0; ni < 4; ++ni) {
            const int oc = n0 + nw * 64 + ni * 16 + lc;
            const float bv = bias[oc];
#pragma unroll
            for (int j = 0; j < 8; ++j) {
                const int pix = m0 + mw * 32 + mi * 16 + lr + j;
                float v = acc[mi][ni][j] + bv;
                if (relu) v = fmaxf(v, 0.0f);
                out[(size_t)pix * outLD + outOff + oc] = f2b(v);
            }
        }
    }
}

// Small heads (OC=18/36): plain VALU 512-MAC dot per output, NCHW f32 out.
__global__ void head_kernel(const __bf16* __restrict__ sim, const float* __restrict__ w,
                            const float* __restrict__ bias, float* __restrict__ out, int OC) {
    size_t i = (size_t)blockIdx.x * blockDim.x + threadIdx.x;
    const size_t total = (size_t)BATCH * OC * PIX;
    if (i >= total) return;
    const int x  = (int)(i & 63);
    const int y  = (int)((i >> 6) % Hh);
    const int oc = (int)((i / PIX) % OC);
    const int n  = (int)(i / ((size_t)PIX * OC));
    const __bf16* a = sim + ((size_t)(n * Hh + y) * Ww + x) * 512;
    const float* wr = w + (size_t)oc * 512;
    float s = bias[oc];
    for (int ic = 0; ic < 512; ++ic) s += (float)a[ic] * wr[ic];
    out[i] = s;
}

// Pairwise softmax: reshape (B,18,H,W)->(B,2,9H,W) pairs channel c with c+9.
__global__ void softmax_kernel(const float* __restrict__ sc, float* __restrict__ prob) {
    size_t i = (size_t)blockIdx.x * blockDim.x + threadIdx.x;
    const size_t total = (size_t)BATCH * 9 * PIX;
    if (i >= total) return;
    const int x = (int)(i & 63);
    const int y = (int)((i >> 6) % Hh);
    const int c = (int)((i / PIX) % 9);
    const int n = (int)(i / ((size_t)PIX * 9));
    const size_t i0 = (((size_t)n * 18 + c) * Hh + y) * Ww + x;
    const size_t i1 = i0 + (size_t)9 * PIX;
    const float s0 = sc[i0], s1 = sc[i1];
    const float m = fmaxf(s0, s1);
    const float e0 = __expf(s0 - m), e1 = __expf(s1 - m);
    const float inv = 1.0f / (e0 + e1);
    prob[i0] = e0 * inv;
    prob[i1] = e1 * inv;
}

// ---------------- launch ----------------
extern "C" void kernel_launch(void* const* d_in, const int* in_sizes, int n_in,
                              void* d_out, int out_size, void* d_ws, size_t ws_size,
                              hipStream_t stream) {
    (void)in_sizes; (void)n_in; (void)out_size; (void)ws_size;
    const float* tf      = (const float*)d_in[0];
    const float* img     = (const float*)d_in[1];
    const float* cc_w    = (const float*)d_in[2];
    const float* cc_bi   = (const float*)d_in[3];
    const float* diff_w  = (const float*)d_in[4];
    const float* diff_bi = (const float*)d_in[5];
    const float* s1_w    = (const float*)d_in[6];
    const float* s1_bi   = (const float*)d_in[7];
    const float* s2_w    = (const float*)d_in[8];
    const float* s2_bi   = (const float*)d_in[9];
    const float* s3_w    = (const float*)d_in[10];
    const float* s3_bi   = (const float*)d_in[11];
    const float* sc_w    = (const float*)d_in[12];
    const float* sc_bi   = (const float*)d_in[13];
    const float* bb_w    = (const float*)d_in[14];
    const float* bb_bi   = (const float*)d_in[15];
    float* out = (float*)d_out;

    // workspace layout (regions aliased across phases)
    char* ws = (char*)d_ws;
    const size_t ACT = (size_t)MTOT * 1024 * sizeof(__bf16);  // 50.3 MB
    float* t1p = (float*)ws;
    float* t2p = (float*)(ws + 16384);
    __bf16* cat = (__bf16*)(ws + 32768);                // img|df_c|cc_c, later sim3
    __bf16* dif = (__bf16*)(ws + 32768 + ACT);          // diffs, later sim1
    __bf16* ccb = (__bf16*)(ws + 32768 + 2 * ACT);      // cc, later sim2
    char* wb = ws + 32768 + 3 * ACT;
    __bf16* wdf = (__bf16*)(wb);                                  // 1024x256
    __bf16* wcc = (__bf16*)(wb + 524288);                         // 1024x256
    __bf16* w1  = (__bf16*)(wb + 2 * 524288);                     // 9216x1024
    __bf16* w2  = (__bf16*)(wb + 2 * 524288 + 18874368);          // 9216x512
    __bf16* w3  = (__bf16*)(wb + 2 * 524288 + 18874368 + 9437184);// 4608x512
    float* score_raw = (float*)(wb + 2 * 524288 + 18874368 + 9437184 + 4718592);
    __bf16* sim1 = dif;   // aliases
    __bf16* sim2 = ccb;
    __bf16* sim3 = cat;

    // 1) target pooling
    pool_kernel<<<(BATCH * 512 + 255) / 256, 256, 0, stream>>>(tf, t1p, t2p);

    // 2) weight transforms (f32 -> WMMA fragment-major bf16)
    wtrans_kernel<<<(256 * 1024 + 255) / 256, 256, 0, stream>>>(diff_w, wdf, 256, 1024, 1);
    wtrans_kernel<<<(256 * 1024 + 255) / 256, 256, 0, stream>>>(cc_w,   wcc, 256, 1024, 1);
    wtrans_kernel<<<(1024 * 1024 * 9 + 255) / 256, 256, 0, stream>>>(s1_w, w1, 1024, 1024, 9);
    wtrans_kernel<<<(512 * 1024 * 9 + 255) / 256, 256, 0, stream>>>(s2_w, w2, 512, 1024, 9);
    wtrans_kernel<<<(512 * 512 * 9 + 255) / 256, 256, 0, stream>>>(s3_w, w3, 512, 512, 9);

    // 3) img transpose + diffs + depthwise 6x6 correlation
    {
        size_t tot = (size_t)BATCH * 512 * PIX;
        corr_diff_kernel<<<(unsigned)((tot + 255) / 256), 256, 0, stream>>>(img, tf, t1p, t2p, cat, dif, ccb);
    }

    // 4) 1x1 convs into cat channels [512:768) and [768:1024)
    conv_gemm_kernel<<<(MTOT / 128) * (256 / 128), 256, 0, stream>>>(dif, wdf, diff_bi, cat, 1024, 256, 0, 1024, 512, 0);
    conv_gemm_kernel<<<(MTOT / 128) * (256 / 128), 256, 0, stream>>>(ccb, wcc, cc_bi,  cat, 1024, 256, 0, 1024, 768, 0);

    // 5) 3x3 conv tower (relu)
    conv_gemm_kernel<<<(MTOT / 128) * (1024 / 128), 256, 0, stream>>>(cat,  w1, s1_bi, sim1, 1024, 1024, 1, 1024, 0, 1);
    conv_gemm_kernel<<<(MTOT / 128) * (512 / 128),  256, 0, stream>>>(sim1, w2, s2_bi, sim2, 1024, 512,  1, 512,  0, 1);
    conv_gemm_kernel<<<(MTOT / 128) * (512 / 128),  256, 0, stream>>>(sim2, w3, s3_bi, sim3, 512,  512,  1, 512,  0, 1);

    // 6) heads
    head_kernel<<<(unsigned)(((size_t)BATCH * 18 * PIX + 255) / 256), 256, 0, stream>>>(sim3, sc_w, sc_bi, score_raw, 18);
    head_kernel<<<(unsigned)(((size_t)BATCH * 36 * PIX + 255) / 256), 256, 0, stream>>>(sim3, bb_w, bb_bi, out + (size_t)BATCH * 18 * PIX, 36);

    // 7) pairwise softmax -> prob
    softmax_kernel<<<(unsigned)(((size_t)BATCH * 9 * PIX + 255) / 256), 256, 0, stream>>>(score_raw, out);
}
